// DiffusionModule_9337258902227
// MI455X (gfx1250) — matrix operand order repaired and checked by reading, not compiled, not run on hardware
//
#include <hip/hip_runtime.h>
#include <hip/hip_bf16.h>

typedef __attribute__((ext_vector_type(16))) __bf16 v16bf;
typedef __attribute__((ext_vector_type(8)))  float  v8f;

#define LDS_STRIDE 40   // 32 data bf16 + 8 pad -> 80B rows, conflict-free b128 frag reads

// ---------------------------------------------------------------------------
// WMMA helper (CDNA5 wave32, V_WMMA_F32_16X16X32_BF16)
// ---------------------------------------------------------------------------
__device__ __forceinline__ v8f wmma_bf16(v16bf a, v16bf b, v8f c) {
    return __builtin_amdgcn_wmma_f32_16x16x32_bf16(
        false, a, false, b, (short)0, c, false, false);
}

// ---------------------------------------------------------------------------
// Async global -> LDS staging of a 128-row x 32-K bf16 tile (8KB of data).
// 256 threads, 2 x 16B chunks each, via GLOBAL_LOAD_ASYNC_TO_LDS_B128
// (ASYNCcnt-tracked; per-lane LDS byte address in VDST, 64-bit vaddr).
// ---------------------------------------------------------------------------
__device__ __forceinline__ void stage_tile_async(const __hip_bfloat16* __restrict__ gbase,
                                                 size_t stride, int row0, int k,
                                                 __hip_bfloat16* lds /* [128][LDS_STRIDE] */) {
    const int t = (int)threadIdx.x;            // 0..255
#pragma unroll
    for (int i = 0; i < 2; ++i) {
        int chunk = t + i * 256;               // 0..511
        int row   = chunk >> 2;                // 0..127
        int c8    = (chunk & 3) * 8;           // element offset in row (0,8,16,24)
        const __hip_bfloat16* g = gbase + (size_t)(row0 + row) * stride + k + c8;
        unsigned l = (unsigned)(uintptr_t)(lds + row * LDS_STRIDE + c8);
        asm volatile("global_load_async_to_lds_b128 %0, %1, off"
                     :: "v"(l), "v"(g) : "memory");
    }
}

__device__ __forceinline__ void wait_async_and_barrier() {
    asm volatile("s_wait_asynccnt 0x0" ::: "memory");
    __syncthreads();
}

// ---------------------------------------------------------------------------
// Fragment reads from LDS.
// A 16x32 bf16: lane&15 = row M; lanes 0-15 K=0..15, lanes 16-31 K=16..31.
// B 32x16 bf16: lane&15 = col N; lanes 0-15 K=0..15, lanes 16-31 K=16..31.
// ---------------------------------------------------------------------------
__device__ __forceinline__ v16bf lds_frag_a(const __hip_bfloat16* tile, int m0) {
    unsigned lane = threadIdx.x & 31u;
    const __hip_bfloat16* p = tile + (m0 + (int)(lane & 15u)) * LDS_STRIDE + (lane >> 4) * 16u;
    union { uint4 u[2]; v16bf v; } t;
    t.u[0] = *reinterpret_cast<const uint4*>(p);
    t.u[1] = *reinterpret_cast<const uint4*>(p + 8);
    return t.v;
}

__device__ __forceinline__ v16bf lds_frag_b(const __hip_bfloat16* tile, int n0) {
    unsigned lane = threadIdx.x & 31u;
    const __hip_bfloat16* p = tile + (n0 + (int)(lane & 15u)) * LDS_STRIDE + (lane >> 4) * 16u;
    union { uint4 u[2]; v16bf v; } t;
    t.u[0] = *reinterpret_cast<const uint4*>(p);
    t.u[1] = *reinterpret_cast<const uint4*>(p + 8);
    return t.v;
}

// ---------------------------------------------------------------------------
// Kernel 0: fp32 [B][C][N] -> bf16 [B][C][N] (srcbf) and bf16 [B][N][C] (srcT)
// ---------------------------------------------------------------------------
__global__ void k_convert_transpose(const float* __restrict__ src,
                                    __hip_bfloat16* __restrict__ srcbf,
                                    __hip_bfloat16* __restrict__ srcT,
                                    int C, int N) {
    __shared__ __hip_bfloat16 tile[32][33];
    const int b  = blockIdx.z;
    const int n0 = blockIdx.x * 32;
    const int c0 = blockIdx.y * 32;
    const float* s = src + (size_t)b * C * N;
#pragma unroll
    for (int i = 0; i < 4; ++i) {
        int c = c0 + (int)threadIdx.y + i * 8;
        int n = n0 + (int)threadIdx.x;
        __hip_bfloat16 h = __float2bfloat16(s[(size_t)c * N + n]);
        srcbf[((size_t)b * C + c) * N + n] = h;
        tile[threadIdx.y + i * 8][threadIdx.x] = h;
    }
    __syncthreads();
#pragma unroll
    for (int i = 0; i < 4; ++i) {
        int n = n0 + (int)threadIdx.y + i * 8;
        int c = c0 + (int)threadIdx.x;
        srcT[((size_t)b * N + n) * C + c] = tile[threadIdx.x][threadIdx.y + i * 8];
    }
}

// ---------------------------------------------------------------------------
// GEMM1: sim[n][m] = scale * sum_c srcT[n][c]*srcT[m][c]
// Block: 128x128 tile, 8 waves of 32(M)x64(N). Double-buffered async LDS.
// Grid: (N/128, N/128).
// ---------------------------------------------------------------------------
__global__ void k_gemm_sim(const __hip_bfloat16* __restrict__ srcT,  // [N][C]
                           float* __restrict__ sim,                  // [N][N]
                           int N, int C, float scale) {
    __shared__ __align__(16) __hip_bfloat16 As[2][128 * LDS_STRIDE];
    __shared__ __align__(16) __hip_bfloat16 Bs[2][128 * LDS_STRIDE];

    const int m0  = blockIdx.x * 128;
    const int n0  = blockIdx.y * 128;
    const int wid = (int)threadIdx.x >> 5;
    const int mw  = (wid >> 1) * 32;
    const int nw  = (wid & 1) * 64;

    stage_tile_async(srcT, (size_t)C, m0, 0, As[0]);
    stage_tile_async(srcT, (size_t)C, n0, 0, Bs[0]);

    v8f acc[2][4] = {};
    const int ksteps = C / 32;
    for (int s = 0; s < ksteps; ++s) {
        const int buf = s & 1;
        wait_async_and_barrier();
        if (s + 1 < ksteps) {
            stage_tile_async(srcT, (size_t)C, m0, (s + 1) * 32, As[buf ^ 1]);
            stage_tile_async(srcT, (size_t)C, n0, (s + 1) * 32, Bs[buf ^ 1]);
        }
        v16bf a0 = lds_frag_a(As[buf], mw);
        v16bf a1 = lds_frag_a(As[buf], mw + 16);
#pragma unroll
        for (int j = 0; j < 4; ++j) {
            v16bf b = lds_frag_b(Bs[buf], nw + 16 * j);
            acc[0][j] = wmma_bf16(a0, b, acc[0][j]);
            acc[1][j] = wmma_bf16(a1, b, acc[1][j]);
        }
    }

    // D layout: VGPR r -> row = r + 8*(lane>=16), col = lane&15
    const unsigned lane  = threadIdx.x & 31u;
    const unsigned col_l = lane & 15u;
    const unsigned row_h = (lane >> 4) * 8u;
#pragma unroll
    for (int mi = 0; mi < 2; ++mi)
#pragma unroll
        for (int j = 0; j < 4; ++j)
#pragma unroll
            for (int r = 0; r < 8; ++r) {
                size_t row = (size_t)(m0 + mw + mi * 16 + (int)row_h + r);
                size_t col = (size_t)(n0 + nw + 16 * j + (int)col_l);
                sim[row * (size_t)N + col] = acc[mi][j][r] * scale;
            }
}

// ---------------------------------------------------------------------------
// Row kernel: mean-subtract, exp(-d^2/2), L1 normalize -> bf16 P[n][m].
// ---------------------------------------------------------------------------
__global__ void k_row_affinity(const float* __restrict__ sim,
                               __hip_bfloat16* __restrict__ P, int N) {
    const int n = blockIdx.x;
    const int t = (int)threadIdx.x;
    const float* row = sim + (size_t)n * N;
    __shared__ float red[256];

    float v[16];
    float s = 0.f;
#pragma unroll
    for (int i = 0; i < 16; ++i) { v[i] = row[t + i * 256]; s += v[i]; }
    red[t] = s; __syncthreads();
    for (int off = 128; off > 0; off >>= 1) {
        if (t < off) red[t] += red[t + off];
        __syncthreads();
    }
    const float mean = red[0] * (1.0f / (float)N);
    __syncthreads();

    float s2 = 0.f;
#pragma unroll
    for (int i = 0; i < 16; ++i) {
        float d = v[i] - mean;
        float a = __expf(-0.5f * d * d);   // SIGMA = 1
        v[i] = a; s2 += a;
    }
    red[t] = s2; __syncthreads();
    for (int off = 128; off > 0; off >>= 1) {
        if (t < off) red[t] += red[t + off];
        __syncthreads();
    }
    const float inv = 1.0f / fmaxf(red[0], 1e-12f);

    __hip_bfloat16* prow = P + (size_t)n * N;
#pragma unroll
    for (int i = 0; i < 16; ++i)
        prow[t + i * 256] = __float2bfloat16(v[i] * inv);
}

// ---------------------------------------------------------------------------
// GEMM2 + blend: out[c][n] = 0.5*(sum_m P[n][m]*srcbf[c][m]) + 0.5*dst[c][n]
// Block: 128(n) x 128(c) tile, double-buffered async LDS, K = N = 4096.
// Epilogue: per lane 8 consecutive n values -> two float4 loads/stores.
// Grid: (N/128, C/128).
// ---------------------------------------------------------------------------
__global__ void k_gemm_diffuse(const __hip_bfloat16* __restrict__ P,     // [N][N]
                               const __hip_bfloat16* __restrict__ srcbf, // [C][N]
                               const float* __restrict__ dst,            // [C][N]
                               float* __restrict__ out,                  // [C][N]
                               int N, int C) {
    __shared__ __align__(16) __hip_bfloat16 As[2][128 * LDS_STRIDE];
    __shared__ __align__(16) __hip_bfloat16 Bs[2][128 * LDS_STRIDE];

    const int m0  = blockIdx.x * 128;     // n dimension
    const int c0  = blockIdx.y * 128;     // c dimension
    const int wid = (int)threadIdx.x >> 5;
    const int mw  = (wid >> 1) * 32;
    const int nw  = (wid & 1) * 64;

    stage_tile_async(P,     (size_t)N, m0, 0, As[0]);
    stage_tile_async(srcbf, (size_t)N, c0, 0, Bs[0]);

    v8f acc[2][4] = {};
    const int ksteps = N / 32;
    for (int s = 0; s < ksteps; ++s) {
        const int buf = s & 1;
        wait_async_and_barrier();
        if (s + 1 < ksteps) {
            stage_tile_async(P,     (size_t)N, m0, (s + 1) * 32, As[buf ^ 1]);
            stage_tile_async(srcbf, (size_t)N, c0, (s + 1) * 32, Bs[buf ^ 1]);
        }
        v16bf a0 = lds_frag_a(As[buf], mw);
        v16bf a1 = lds_frag_a(As[buf], mw + 16);
#pragma unroll
        for (int j = 0; j < 4; ++j) {
            v16bf b = lds_frag_b(Bs[buf], nw + 16 * j);
            acc[0][j] = wmma_bf16(a0, b, acc[0][j]);
            acc[1][j] = wmma_bf16(a1, b, acc[1][j]);
        }
    }

    const unsigned lane  = threadIdx.x & 31u;
    const unsigned col_l = lane & 15u;
    const unsigned row_h = (lane >> 4) * 8u;
#pragma unroll
    for (int mi = 0; mi < 2; ++mi)
#pragma unroll
        for (int j = 0; j < 4; ++j) {
            const int nn = m0 + mw + mi * 16 + (int)row_h;   // base n (8 consecutive)
            const int cc = c0 + nw + 16 * j + (int)col_l;    // c index
            const size_t idx = (size_t)cc * N + nn;          // 16B aligned (nn % 8 == 0)
            float4 d0 = *reinterpret_cast<const float4*>(dst + idx);
            float4 d1 = *reinterpret_cast<const float4*>(dst + idx + 4);
            float4 o0, o1;
            o0.x = 0.5f * acc[mi][j][0] + 0.5f * d0.x;
            o0.y = 0.5f * acc[mi][j][1] + 0.5f * d0.y;
            o0.z = 0.5f * acc[mi][j][2] + 0.5f * d0.z;
            o0.w = 0.5f * acc[mi][j][3] + 0.5f * d0.w;
            o1.x = 0.5f * acc[mi][j][4] + 0.5f * d1.x;
            o1.y = 0.5f * acc[mi][j][5] + 0.5f * d1.y;
            o1.z = 0.5f * acc[mi][j][6] + 0.5f * d1.z;
            o1.w = 0.5f * acc[mi][j][7] + 0.5f * d1.w;
            *reinterpret_cast<float4*>(out + idx)     = o0;
            *reinterpret_cast<float4*>(out + idx + 4) = o1;
        }
}

// ---------------------------------------------------------------------------
// Launch: convert once, then per batch: GEMM1 -> row affinity -> GEMM2+blend
// Workspace: srcT (8Mi) | srcbf (8Mi) | sim fp32 (64Mi, reused) | P bf16 (32Mi)
// ---------------------------------------------------------------------------
extern "C" void kernel_launch(void* const* d_in, const int* in_sizes, int n_in,
                              void* d_out, int out_size, void* d_ws, size_t ws_size,
                              hipStream_t stream) {
    (void)in_sizes; (void)n_in; (void)out_size; (void)ws_size;
    const int B = 4, C = 256, H = 64, W = 64;
    const int N = H * W;                       // 4096

    const float* src = (const float*)d_in[0];
    const float* dst = (const float*)d_in[1];
    float* out = (float*)d_out;

    char* ws = (char*)d_ws;
    __hip_bfloat16* srcT  = (__hip_bfloat16*)ws;                                 // [B][N][C]
    __hip_bfloat16* srcbf = (__hip_bfloat16*)(ws + (size_t)B * N * C * 2);       // [B][C][N]
    float*          sim   = (float*)(ws + (size_t)2 * B * N * C * 2);            // [N][N]
    __hip_bfloat16* P     = (__hip_bfloat16*)(ws + (size_t)2 * B * N * C * 2
                                                 + (size_t)N * N * 4);           // [N][N]

    dim3 blkT(32, 8, 1), grdT(N / 32, C / 32, B);
    k_convert_transpose<<<grdT, blkT, 0, stream>>>(src, srcbf, srcT, C, N);

    const float scale = 0.0625f;               // 1/sqrt(256)
    dim3 g1(N / 128, N / 128, 1);              // 32 x 32 blocks
    dim3 g2(N / 128, C / 128, 1);              // 32 x 2 blocks

    for (int b = 0; b < B; ++b) {
        const __hip_bfloat16* srcT_b  = srcT  + (size_t)b * N * C;
        const __hip_bfloat16* srcbf_b = srcbf + (size_t)b * C * N;
        const float* dst_b = dst + (size_t)b * C * N;
        float*       out_b = out + (size_t)b * C * N;

        k_gemm_sim    <<<g1, 256, 0, stream>>>(srcT_b, sim, N, C, scale);
        k_row_affinity<<<N,  256, 0, stream>>>(sim, P, N);
        k_gemm_diffuse<<<g2, 256, 0, stream>>>(P, srcbf_b, dst_b, out_b, N, C);
    }
}
